// SimpleHGN_MTL_18580028522790
// MI455X (gfx1250) — matrix operand order repaired and checked by reading, not compiled, not run on hardware
//
#include <hip/hip_runtime.h>
#include <hip/hip_bf16.h>

// ---------------------------------------------------------------------------
// SimpleHGN forward for MI455X (gfx1250, wave32, WMMA).
// Dense projections run on v_wmma_f32_16x16x32_f16; sparse attention phase is
// HBM/atomic bound and uses f16-compressed projected features to halve gather
// traffic.
// ---------------------------------------------------------------------------

#define NH_C    50000
#define NT_C    100000
#define NE_C    800000
#define F_IN_C  256
#define D_OUT_C 32
#define H_C     8
#define E_FEAT_C 64
#define HD_C    (D_OUT_C * H_C)   // 256 output cols
#define NEG_SLOPE_C 0.2f

typedef __attribute__((ext_vector_type(16))) _Float16 v16h;
typedef __attribute__((ext_vector_type(8)))  float    v8f;
typedef __attribute__((ext_vector_type(4)))  float    v4f;

// ---------------- utility: zero fill (graph-capture-safe memset) -----------
__global__ void zero_f32_kernel(float* __restrict__ p, long long n) {
  long long i = (long long)blockIdx.x * blockDim.x + threadIdx.x;
  if (i < n) p[i] = 0.0f;
}

// ---------------- pack W (f32, K x N row-major) into WMMA B fragments ------
// Packed layout: Bp[(((kc*16)+ntile)*32 + lane)*16 + j] (f16, 32B per lane)
// B fragment (32x16 f16): lanes 0-15 hold K = kb+0..7,16..23,
// lanes 16-31 hold K = kb+8..15,24..31, column N = ntile*16 + (lane&15).
__global__ void pack_w_kernel(const float* __restrict__ W, _Float16* __restrict__ Bp) {
  int idx = blockIdx.x * blockDim.x + threadIdx.x;
  if (idx >= F_IN_C * HD_C) return;
  int j    = idx & 15;
  int lane = (idx >> 4) & 31;
  int nt   = (idx >> 9) & 15;
  int kc   = idx >> 13;
  int k = kc * 32 + ((lane >> 4) ? 8 : 0) + (j < 8 ? j : j + 8);
  int n = nt * 16 + (lane & 15);
  Bp[idx] = (_Float16)W[k * HD_C + n];
}

// ---------------- WMMA GEMM: C[M x 256] = A[M x 256] @ W ------------------
// Block = 128 threads (4 waves). Block handles 16 rows x 256 cols.
// Wave w handles n-tiles [4w, 4w+4). 8 k-chunks of 32 -> 32 v_wmma per wave.
__global__ __launch_bounds__(128) void gemm_wmma_f16_kernel(
    const float* __restrict__ A, const v16h* __restrict__ Bp,
    _Float16* __restrict__ C, int mrows) {
  const int wave = threadIdx.x >> 5;
  const int lane = threadIdx.x & 31;
  const int half = lane >> 4;   // which half-wave
  const int l16  = lane & 15;
  const int m0   = blockIdx.x * 16;
  if (m0 >= mrows) return;

  const float* arow = A + (size_t)(m0 + l16) * F_IN_C;

  v8f acc0 = {}, acc1 = {}, acc2 = {}, acc3 = {};

#pragma unroll
  for (int kc = 0; kc < 8; ++kc) {
    // A fragment, documented 16-bit A layout:
    // lane<16 : elems 0..7 = K kb+0..7,  elems 8..15 = K kb+16..23
    // lane>=16: elems 0..7 = K kb+8..15, elems 8..15 = K kb+24..31
    const int kb = kc * 32 + half * 8;
    v4f a0 = *(const v4f*)(arow + kb);
    v4f a1 = *(const v4f*)(arow + kb + 4);
    v4f a2 = *(const v4f*)(arow + kb + 16);
    v4f a3 = *(const v4f*)(arow + kb + 20);
    v16h af;
#pragma unroll
    for (int j = 0; j < 4; ++j) {
      af[j]      = (_Float16)a0[j];
      af[4 + j]  = (_Float16)a1[j];
      af[8 + j]  = (_Float16)a2[j];
      af[12 + j] = (_Float16)a3[j];
    }
    const v16h* bbase = Bp + ((kc * 16 + wave * 4) * 32) + lane;
    v16h b0 = bbase[0];
    v16h b1 = bbase[32];
    v16h b2 = bbase[64];
    v16h b3 = bbase[96];
    acc0 = __builtin_amdgcn_wmma_f32_16x16x32_f16(false, af, false, b0, (short)0, acc0, false, false);
    acc1 = __builtin_amdgcn_wmma_f32_16x16x32_f16(false, af, false, b1, (short)0, acc1, false, false);
    acc2 = __builtin_amdgcn_wmma_f32_16x16x32_f16(false, af, false, b2, (short)0, acc2, false, false);
    acc3 = __builtin_amdgcn_wmma_f32_16x16x32_f16(false, af, false, b3, (short)0, acc3, false, false);
  }

  // C/D layout: elem v of v8f -> row m0 + v + 8*half, col ntile*16 + l16.
  v8f accs[4] = {acc0, acc1, acc2, acc3};
#pragma unroll
  for (int nt = 0; nt < 4; ++nt) {
    const int ncol = (wave * 4 + nt) * 16 + l16;
#pragma unroll
    for (int v = 0; v < 8; ++v) {
      const int mrow = m0 + v + half * 8;
      C[(size_t)mrow * HD_C + ncol] = (_Float16)accs[nt][v];
    }
  }
}

// ---------------- per-node attention half-logit: sum_d a[h][d]*Hf[n][h][d] -
__global__ void node_logits_kernel(const _Float16* __restrict__ Hf,
                                   const float* __restrict__ a,
                                   float* __restrict__ outl, int nnodes) {
  int t = blockIdx.x * blockDim.x + threadIdx.x;
  if (t >= nnodes * H_C) return;
  const int node = t >> 3, h = t & 7;
  const _Float16* p = Hf + (size_t)node * HD_C + h * D_OUT_C;
  const float* av = a + h * D_OUT_C;
  float s = 0.0f;
#pragma unroll
  for (int d = 0; d < D_OUT_C; ++d) s += av[d] * (float)p[d];
  outl[t] = s;
}

// ---------------- edge-type half-logit (tiny: R=5 x H=8) ------------------
__global__ void he_kernel(const float* __restrict__ edge_emb,
                          const float* __restrict__ W_e,
                          const float* __restrict__ a_e,
                          float* __restrict__ he) {
  int t = threadIdx.x;
  if (t >= 5 * H_C) return;
  const int r = t >> 3, h = t & 7;
  float s = 0.0f;
  for (int j = 0; j < E_FEAT_C; ++j) {
    float ej = 0.0f;
    for (int k = 0; k < E_FEAT_C; ++k)
      ej += edge_emb[r * E_FEAT_C + k] * W_e[k * (E_FEAT_C * H_C) + h * E_FEAT_C + j];
    s += a_e[h * E_FEAT_C + j] * ej;
  }
  he[t] = s;
}

// ---------------- ordered-uint encoding for float atomic max ---------------
__device__ __forceinline__ unsigned enc_f32(float f) {
  unsigned u = __float_as_uint(f);
  return (u & 0x80000000u) ? ~u : (u | 0x80000000u);
}
__device__ __forceinline__ float dec_f32(unsigned e) {
  unsigned u = (e & 0x80000000u) ? (e & 0x7fffffffu) : ~e;
  return __uint_as_float(u);
}

// ---------------- edge pass 1: logits + segment max ------------------------
__global__ void edge_att_kernel(const int* __restrict__ head,
                                const int* __restrict__ tail,
                                const int* __restrict__ etype,
                                const float* __restrict__ hl,
                                const float* __restrict__ hr,
                                const float* __restrict__ he,
                                float* __restrict__ att,
                                unsigned* __restrict__ segmax) {
  int t = blockIdx.x * blockDim.x + threadIdx.x;
  if (t >= NE_C * H_C) return;
  const int e = t >> 3, h = t & 7;
  const int hd = head[e];
  float x = hl[hd * H_C + h] + hr[tail[e] * H_C + h] + he[etype[e] * H_C + h];
  x = (x > 0.0f) ? x : NEG_SLOPE_C * x;
  att[t] = x;
  atomicMax(&segmax[hd * H_C + h], enc_f32(x));
}

// ---------------- edge pass 2: exp + segment denom -------------------------
__global__ void edge_exp_kernel(const int* __restrict__ head,
                                float* __restrict__ att,       // in: logit, out: exp
                                const unsigned* __restrict__ segmax,
                                float* __restrict__ denom) {
  int t = blockIdx.x * blockDim.x + threadIdx.x;
  if (t >= NE_C * H_C) return;
  const int e = t >> 3, h = t & 7;
  const int hd = head[e];
  float ex = __expf(att[t] - dec_f32(segmax[hd * H_C + h]));
  att[t] = ex;
  atomicAdd(&denom[hd * H_C + h], ex);
}

// ---------------- aggregation: out[head] += alpha * h_tail[tail] -----------
#define EDGES_PER_BLOCK 8
__global__ __launch_bounds__(256) void aggregate_kernel(
    const int* __restrict__ head, const int* __restrict__ tail,
    const float* __restrict__ ex, const float* __restrict__ denom,
    const _Float16* __restrict__ Ht, float* __restrict__ out) {
  const int f = threadIdx.x;       // feature 0..255
  const int h = f >> 5;            // head index of this feature
  const int e0 = blockIdx.x * EDGES_PER_BLOCK;
#pragma unroll
  for (int i = 0; i < EDGES_PER_BLOCK; ++i) {
    const int e = e0 + i;
    if (e >= NE_C) return;
    const int hd = head[e], tl = tail[e];
    const float alpha = ex[e * H_C + h] / denom[hd * H_C + h];
    const float val = alpha * (float)Ht[(size_t)tl * HD_C + f];
    atomicAdd(&out[(size_t)hd * HD_C + f], val);
  }
}

// ---------------- in-place ELU ---------------------------------------------
__global__ void elu_kernel(float* __restrict__ out, long long n) {
  long long i = (long long)blockIdx.x * blockDim.x + threadIdx.x;
  if (i >= n) return;
  float x = out[i];
  out[i] = (x > 0.0f) ? x : (__expf(x) - 1.0f);
}

// ---------------------------------------------------------------------------
extern "C" void kernel_launch(void* const* d_in, const int* in_sizes, int n_in,
                              void* d_out, int out_size, void* d_ws, size_t ws_size,
                              hipStream_t stream) {
  (void)in_sizes; (void)n_in; (void)out_size; (void)ws_size;

  const float* head_feature = (const float*)d_in[0];
  const float* tail_feature = (const float*)d_in[1];
  const int*   head_ind     = (const int*)d_in[2];
  const int*   tail_ind     = (const int*)d_in[3];
  const int*   tmp_edge     = (const int*)d_in[4];
  const float* W            = (const float*)d_in[5];
  const float* W_e          = (const float*)d_in[6];
  const float* a_l          = (const float*)d_in[7];
  const float* a_r          = (const float*)d_in[8];
  const float* a_e          = (const float*)d_in[9];
  const float* edge_emb     = (const float*)d_in[10];
  float* out = (float*)d_out;

  // ---- workspace carve-out (all offsets 256B aligned) ----
  char* ws = (char*)d_ws;
  size_t off = 0;
  auto take = [&](size_t bytes) { void* p = ws + off; off += (bytes + 255) & ~(size_t)255; return p; };
  _Float16* Bp     = (_Float16*)take((size_t)F_IN_C * HD_C * sizeof(_Float16));   // 128 KB
  _Float16* Hh     = (_Float16*)take((size_t)NH_C * HD_C * sizeof(_Float16));     // 25.6 MB
  _Float16* Ht     = (_Float16*)take((size_t)NT_C * HD_C * sizeof(_Float16));     // 51.2 MB
  float*    hl     = (float*)take((size_t)NH_C * H_C * sizeof(float));            // 1.6 MB
  float*    hr     = (float*)take((size_t)NT_C * H_C * sizeof(float));            // 3.2 MB
  float*    he     = (float*)take((size_t)5 * H_C * sizeof(float));
  float*    att    = (float*)take((size_t)NE_C * H_C * sizeof(float));            // 25.6 MB
  unsigned* segmax = (unsigned*)take((size_t)NH_C * H_C * sizeof(unsigned));      // 1.6 MB
  float*    denom  = (float*)take((size_t)NH_C * H_C * sizeof(float));            // 1.6 MB

  const long long nout = (long long)NH_C * HD_C;

  // 1) zero accumulators (segmax encoding: 0 < enc(x) for all finite x)
  zero_f32_kernel<<<(unsigned)((nout + 255) / 256), 256, 0, stream>>>(out, nout);
  zero_f32_kernel<<<(NH_C * H_C + 255) / 256, 256, 0, stream>>>(denom, NH_C * H_C);
  zero_f32_kernel<<<(NH_C * H_C + 255) / 256, 256, 0, stream>>>((float*)segmax, NH_C * H_C);

  // 2) pack W into f16 WMMA B fragments
  pack_w_kernel<<<(F_IN_C * HD_C + 255) / 256, 256, 0, stream>>>(W, Bp);

  // 3) projections via WMMA
  gemm_wmma_f16_kernel<<<NH_C / 16, 128, 0, stream>>>(head_feature, (const v16h*)Bp, Hh, NH_C);
  gemm_wmma_f16_kernel<<<NT_C / 16, 128, 0, stream>>>(tail_feature, (const v16h*)Bp, Ht, NT_C);

  // 4) attention half-logits
  he_kernel<<<1, 64, 0, stream>>>(edge_emb, W_e, a_e, he);
  node_logits_kernel<<<(NH_C * H_C + 255) / 256, 256, 0, stream>>>(Hh, a_l, hl, NH_C);
  node_logits_kernel<<<(NT_C * H_C + 255) / 256, 256, 0, stream>>>(Ht, a_r, hr, NT_C);

  // 5) edge softmax (segment max -> exp -> segment sum)
  edge_att_kernel<<<(NE_C * H_C + 255) / 256, 256, 0, stream>>>(
      head_ind, tail_ind, tmp_edge, hl, hr, he, att, segmax);
  edge_exp_kernel<<<(NE_C * H_C + 255) / 256, 256, 0, stream>>>(
      head_ind, att, segmax, denom);

  // 6) weighted scatter aggregation into d_out
  aggregate_kernel<<<(NE_C + EDGES_PER_BLOCK - 1) / EDGES_PER_BLOCK, 256, 0, stream>>>(
      head_ind, tail_ind, att, denom, Ht, out);

  // 7) ELU in place
  elu_kernel<<<(unsigned)((nout + 255) / 256), 256, 0, stream>>>(out, nout);
}
